// KoLeoLoss_13924283973723
// MI455X (gfx1250) — matrix operand order repaired and checked by reading, not compiled, not run on hardware
//
#include <hip/hip_runtime.h>
#include <hip/hip_fp16.h>

typedef __attribute__((ext_vector_type(16))) _Float16 v16h;
typedef __attribute__((ext_vector_type(8)))  _Float16 v8h;
typedef __attribute__((ext_vector_type(8)))  float    v8f;
typedef __attribute__((ext_vector_type(4)))  int      v4i;

#define B_ROWS 16384
#define DIMK   1024
#define TILE   128      // block output tile: 128x128
#define KCHUNK 32       // K per WMMA step
#define NCHUNK (DIMK / KCHUNK)
#define LDS_STRIDE 40   // halves per row in LDS (32 + 8 pad), 80B, 16B-aligned

#if defined(__AMDGCN__) && __has_builtin(__builtin_amdgcn_global_load_async_to_lds_b128)
#define USE_ASYNC 1
#else
#define USE_ASYNC 0
#endif

// ---------------------------------------------------------------------------
// Kernel 0: reset row-max keys (ws is poisoned; must be deterministic per call)
// ---------------------------------------------------------------------------
__global__ void koleo_init_keys(unsigned* __restrict__ keys) {
    int i = blockIdx.x * blockDim.x + threadIdx.x;
    if (i < B_ROWS) keys[i] = 0u;   // 0 maps below -inf in the monotonic key space
}

// ---------------------------------------------------------------------------
// Kernel 1: L2-normalize each row, emit f16 copy for the WMMA GEMM
// ---------------------------------------------------------------------------
__global__ __launch_bounds__(256) void koleo_normalize(const float* __restrict__ x,
                                                       _Float16* __restrict__ xn) {
    const int row = blockIdx.x;
    const float* xr = x + (size_t)row * DIMK;
    float s = 0.f;
    for (int i = threadIdx.x; i < DIMK; i += 256) { float v = xr[i]; s += v * v; }
    __shared__ float red[256];
    red[threadIdx.x] = s;
    __syncthreads();
    for (int off = 128; off > 0; off >>= 1) {
        if (threadIdx.x < off) red[threadIdx.x] += red[threadIdx.x + off];
        __syncthreads();
    }
    const float inv = rsqrtf(fmaxf(red[0], 1e-24f));   // norm clamp 1e-12 -> sumsq 1e-24
    _Float16* o = xn + (size_t)row * DIMK;
    for (int i = threadIdx.x; i < DIMK; i += 256) o[i] = (_Float16)(xr[i] * inv);
}

// ---------------------------------------------------------------------------
// Kernel 2: tiled Gram-matrix with per-row running max (diagonal masked)
// Each wave: 32 rows x 64 cols (2 A-frags x 4 B-frags, 8 accumulators).
// ---------------------------------------------------------------------------
__device__ inline v16h load_frag_2x8(const _Float16* lo, const _Float16* hi) {
    v16h r;
    *(v8h*)&r       = *(const v8h*)lo;
    *((v8h*)&r + 1) = *(const v8h*)hi;
    return r;
}

#define WMMA_F16(af, bfv, accv) \
    __builtin_amdgcn_wmma_f32_16x16x32_f16(false, (af), false, (bfv), (short)0, (accv), false, false)

__global__ __launch_bounds__(256, 2) void koleo_simmax(const _Float16* __restrict__ xn,
                                                       unsigned* __restrict__ keys) {
    __shared__ _Float16 ldsA[2][TILE * LDS_STRIDE];
    __shared__ _Float16 ldsB[2][TILE * LDS_STRIDE];

    const int rowBase = blockIdx.x * TILE;
    const int colBase = blockIdx.y * TILE;
    const int tid  = threadIdx.x;
    const int wave = tid >> 5;
    const int lane = tid & 31;
    const int nloc = lane & 15;
    const bool hiHalf = (lane >= 16);

    // Per-thread staging geometry: 2 x 16B chunks per matrix per chunk
    const int c0 = tid,        c1 = tid + 256;
    const int r0 = c0 >> 2,    k80 = (c0 & 3) * 8;
    const int r1 = c1 >> 2,    k81 = (c1 & 3) * 8;
    const _Float16* gA0 = &xn[(size_t)(rowBase + r0) * DIMK + k80];
    const _Float16* gA1 = &xn[(size_t)(rowBase + r1) * DIMK + k81];
    const _Float16* gB0 = &xn[(size_t)(colBase + r0) * DIMK + k80];
    const _Float16* gB1 = &xn[(size_t)(colBase + r1) * DIMK + k81];
    const int dA0 = r0 * LDS_STRIDE + k80, dA1 = r1 * LDS_STRIDE + k81;

    // 8 named accumulators: acc<a><t>, a = row-tile (0..1), t = col-tile (0..3)
    v8f acc00 = {}, acc01 = {}, acc02 = {}, acc03 = {};
    v8f acc10 = {}, acc11 = {}, acc12 = {}, acc13 = {};

    const int rowGrp = wave >> 1;           // 0..3  -> 32-row strip
    const int colGrp = wave & 1;            // 0..1  -> 64-col strip
    const int mrow0 = (rowGrp << 5) + nloc;         // A-frag rows (tile a=0)
    const int mrow1 = mrow0 + 16;                   // A-frag rows (tile a=1)
    const int aOff = hiHalf ? 8 : 0;        // A 16x32: lane<16 K{0..7,16..23}
    const int bOff = hiHalf ? 8 : 0;        // B 32x16: lane<16 K{0..7,16..23}
    const int ncolBase = (colGrp << 6) + nloc;

    auto compute = [&](const _Float16* cA, const _Float16* cB) {
        const v16h af0 = load_frag_2x8(&cA[mrow0 * LDS_STRIDE + aOff],
                                       &cA[mrow0 * LDS_STRIDE + aOff + 16]);
        const v16h af1 = load_frag_2x8(&cA[mrow1 * LDS_STRIDE + aOff],
                                       &cA[mrow1 * LDS_STRIDE + aOff + 16]);
        #define LDB(t) load_frag_2x8(&cB[(ncolBase + ((t) << 4)) * LDS_STRIDE + bOff], \
                                     &cB[(ncolBase + ((t) << 4)) * LDS_STRIDE + bOff + 16])
        {
            const v16h b0 = LDB(0), b1 = LDB(1);
            acc00 = WMMA_F16(af0, b0, acc00);
            acc10 = WMMA_F16(af1, b0, acc10);
            acc01 = WMMA_F16(af0, b1, acc01);
            acc11 = WMMA_F16(af1, b1, acc11);
        }
        {
            const v16h b2 = LDB(2), b3 = LDB(3);
            acc02 = WMMA_F16(af0, b2, acc02);
            acc12 = WMMA_F16(af1, b2, acc12);
            acc03 = WMMA_F16(af0, b3, acc03);
            acc13 = WMMA_F16(af1, b3, acc13);
        }
        #undef LDB
    };

    // Pointer-swap double buffering (no unroll -> one loop body, stable acc regs)
    const _Float16* curA = ldsA[0];
    const _Float16* curB = ldsB[0];
    _Float16* nxtA = ldsA[1];
    _Float16* nxtB = ldsB[1];

#if USE_ASYNC
    auto stage = [&](_Float16* tA, _Float16* tB, int kc) {
        __builtin_amdgcn_global_load_async_to_lds_b128(
            (v4i*)(void*)(gA0 + kc), (v4i*)(void*)(tA + dA0), 0, 0);
        __builtin_amdgcn_global_load_async_to_lds_b128(
            (v4i*)(void*)(gA1 + kc), (v4i*)(void*)(tA + dA1), 0, 0);
        __builtin_amdgcn_global_load_async_to_lds_b128(
            (v4i*)(void*)(gB0 + kc), (v4i*)(void*)(tB + dA0), 0, 0);
        __builtin_amdgcn_global_load_async_to_lds_b128(
            (v4i*)(void*)(gB1 + kc), (v4i*)(void*)(tB + dA1), 0, 0);
    };
    stage((_Float16*)curA, (_Float16*)curB, 0);
    asm volatile("s_wait_asynccnt 0" ::: "memory");
    __syncthreads();
    #pragma clang loop unroll(disable)
    for (int j = 0; j < NCHUNK - 1; ++j) {
        stage(nxtA, nxtB, (j + 1) * KCHUNK);   // in flight during compute
        compute(curA, curB);
        asm volatile("s_wait_asynccnt 0" ::: "memory");
        __syncthreads();
        _Float16* tA = (_Float16*)curA; curA = nxtA; nxtA = tA;
        _Float16* tB = (_Float16*)curB; curB = nxtB; nxtB = tB;
    }
    compute(curA, curB);
#else
    v8h sA0, sA1, sB0, sB1;
    auto stage_load = [&](int kc) {
        sA0 = *(const v8h*)(gA0 + kc);
        sA1 = *(const v8h*)(gA1 + kc);
        sB0 = *(const v8h*)(gB0 + kc);
        sB1 = *(const v8h*)(gB1 + kc);
    };
    auto stage_store = [&](_Float16* tA, _Float16* tB) {
        *(v8h*)(tA + dA0) = sA0;
        *(v8h*)(tA + dA1) = sA1;
        *(v8h*)(tB + dA0) = sB0;
        *(v8h*)(tB + dA1) = sB1;
    };
    stage_load(0);
    stage_store((_Float16*)curA, (_Float16*)curB);
    __syncthreads();
    #pragma clang loop unroll(disable)
    for (int j = 0; j < NCHUNK - 1; ++j) {
        stage_load((j + 1) * KCHUNK);          // in flight during compute
        compute(curA, curB);
        stage_store(nxtA, nxtB);
        __syncthreads();
        _Float16* tA = (_Float16*)curA; curA = nxtA; nxtA = tA;
        _Float16* tB = (_Float16*)curB; curB = nxtB; nxtB = tB;
    }
    compute(curA, curB);
#endif

    // C/D layout: acc[a][t][v] at lane L holds M = v + (L>=16 ? 8 : 0), N = t*16 + (L&15)
    const v8f accArr[2][4] = {{acc00, acc01, acc02, acc03},
                              {acc10, acc11, acc12, acc13}};
    const float NEG_INF = -__builtin_inff();
    const int halfOff = hiHalf ? 8 : 0;
    const int waveRow = rowBase + (rowGrp << 5);
    const int waveCol = colBase + (colGrp << 6);
    #pragma unroll
    for (int a = 0; a < 2; ++a) {
        #pragma unroll
        for (int v = 0; v < 8; ++v) {
            const int grow = waveRow + (a << 4) + v + halfOff;
            float m = NEG_INF;
            #pragma unroll
            for (int t = 0; t < 4; ++t) {
                const int gcol = waveCol + (t << 4) + nloc;
                float val = accArr[a][t][v];
                if (gcol == grow) val = NEG_INF;   // exclude self-similarity
                m = fmaxf(m, val);
            }
            #pragma unroll
            for (int mask = 1; mask < 16; mask <<= 1)
                m = fmaxf(m, __shfl_xor(m, mask, 32));
            if (nloc == 0) {
                const unsigned u = __float_as_uint(m);
                const unsigned key = (u & 0x80000000u) ? ~u : (u | 0x80000000u);
                atomicMax(&keys[grow], key);
            }
        }
    }
}

// ---------------------------------------------------------------------------
// Kernel 3: decode keys, loss = -mean(0.5 * log(2 - 2*max + eps))
// ---------------------------------------------------------------------------
__global__ __launch_bounds__(1024) void koleo_finalize(const unsigned* __restrict__ keys,
                                                       float* __restrict__ out) {
    __shared__ float red[1024];
    float s = 0.f;
    for (int i = threadIdx.x; i < B_ROWS; i += 1024) {
        const unsigned key = keys[i];
        const unsigned u = (key & 0x80000000u) ? (key & 0x7FFFFFFFu) : ~key;
        const float mx = __uint_as_float(u);
        const float d2 = 2.0f - 2.0f * mx + 1e-8f;
        s += 0.5f * logf(d2);
    }
    red[threadIdx.x] = s;
    __syncthreads();
    for (int off = 512; off > 0; off >>= 1) {
        if (threadIdx.x < off) red[threadIdx.x] += red[threadIdx.x + off];
        __syncthreads();
    }
    if (threadIdx.x == 0) out[0] = -red[0] / (float)B_ROWS;
}

// ---------------------------------------------------------------------------
extern "C" void kernel_launch(void* const* d_in, const int* in_sizes, int n_in,
                              void* d_out, int out_size, void* d_ws, size_t ws_size,
                              hipStream_t stream) {
    (void)in_sizes; (void)n_in; (void)out_size; (void)ws_size;
    const float* x = (const float*)d_in[0];
    float* out = (float*)d_out;

    _Float16* xn = (_Float16*)d_ws;                                        // 32 MB
    unsigned* keys = (unsigned*)((char*)d_ws +
                                 (size_t)B_ROWS * DIMK * sizeof(_Float16)); // 64 KB

    koleo_init_keys<<<(B_ROWS + 255) / 256, 256, 0, stream>>>(keys);
    koleo_normalize<<<B_ROWS, 256, 0, stream>>>(x, xn);
    dim3 grid(B_ROWS / TILE, B_ROWS / TILE);   // 128 x 128 blocks
    koleo_simmax<<<grid, 256, 0, stream>>>(xn, keys);
    koleo_finalize<<<1, 1024, 0, stream>>>(keys, out);
}
